// AttentionModel_53558242181838
// MI455X (gfx1250) — matrix operand order repaired and checked by reading, not compiled, not run on hardware
//
#include <hip/hip_runtime.h>

// ---------------------------------------------------------------------------
// Types for CDNA5 WMMA (wave32, 16x16x32 bf16 -> f32)
// ---------------------------------------------------------------------------
typedef __bf16 v8bf  __attribute__((ext_vector_type(8)));
typedef __bf16 v16bf __attribute__((ext_vector_type(16)));
typedef float  v8f   __attribute__((ext_vector_type(8)));

#define WMMA_BF16(a, b, c) \
  __builtin_amdgcn_wmma_f32_16x16x32_bf16(false, (a), false, (b), (short)0, (c), false, false)

union BF16x16 { v16bf v; v8bf h[2]; };

// A-matrix 16x32 bf16 fragment (ISA 7.12.2): lane l (l16=l%16, sel=l/16):
//   holds row M=l16, K = {sel*8..sel*8+7} and {16+sel*8..16+sel*8+7}
__device__ __forceinline__ v16bf frag_a(const __bf16* rowp, int sel) {
  BF16x16 u;
  u.h[0] = *(const v8bf*)(rowp + sel * 8);
  u.h[1] = *(const v8bf*)(rowp + sel * 8 + 16);
  return u.v;
}
// B-matrix 32x16 bf16 fragment: lane holds column N=l16, K = {sel*8..sel*8+7} and {16+sel*8..16+sel*8+7}
// (B supplied K-contiguous per column, i.e. Bt[N][K] rows)
__device__ __forceinline__ v16bf frag_b(const __bf16* colp, int sel) {
  BF16x16 u;
  u.h[0] = *(const v8bf*)(colp + sel * 8);
  u.h[1] = *(const v8bf*)(colp + sel * 8 + 16);
  return u.v;
}

// Cross-lane xor-butterfly within 16-lane rows via DPP16 ROW_XMASK (0x160|mask):
// pure VALU, no LDS round-trip, no dscnt waits.
template <int MASK>
__device__ __forceinline__ float xor16f(float x) {
  union { float f; int i; } u, r;
  u.f = x;
  r.i = __builtin_amdgcn_update_dpp(0, u.i, 0x160 + MASK, 0xf, 0xf, true);
  return r.f;
}
__device__ __forceinline__ float redmax16(float x) {
  x = fmaxf(x, xor16f<1>(x));
  x = fmaxf(x, xor16f<2>(x));
  x = fmaxf(x, xor16f<4>(x));
  x = fmaxf(x, xor16f<8>(x));
  return x;
}
__device__ __forceinline__ float redsum16(float x) {
  x += xor16f<1>(x);
  x += xor16f<2>(x);
  x += xor16f<4>(x);
  x += xor16f<8>(x);
  return x;
}

// ---------------------------------------------------------------------------
// Async global->LDS DMA (GLOBAL_LOAD_ASYNC_TO_LDS_B128, tracked by ASYNCcnt).
// LDS operand is the 32-bit LDS byte offset = integer value of the AS(3) ptr.
// ---------------------------------------------------------------------------
__device__ __forceinline__ void async_copy_b128(const __bf16* gsrc, __bf16* ldst) {
  unsigned loff =
      (unsigned)(unsigned long long)(__attribute__((address_space(3))) const __bf16*)ldst;
  asm volatile("global_load_async_to_lds_b128 %0, %1, off"
               :
               : "v"(loff), "v"(gsrc)
               : "memory");
}
__device__ __forceinline__ void wait_async0() {
  asm volatile("s_wait_asynccnt 0x0" ::: "memory");
}

// ---------------------------------------------------------------------------
// Model constants
// ---------------------------------------------------------------------------
#define LYR   4
#define DMODEL 512
#define NHEAD 8
#define DFFN  2048
#define DIN_  64
#define DOUT_ 10
#define DK_   64
#define SEQ   2048
#define NBATCH 4
#define NTOK  (NBATCH * SEQ)   // 8192

// ---------------------------------------------------------------------------
// Elementwise f32 -> bf16
// ---------------------------------------------------------------------------
__global__ void cvt_bf16_kernel(const float* __restrict__ in, __bf16* __restrict__ out, int n) {
  int i = blockIdx.x * 256 + threadIdx.x;
  if (i < n) out[i] = (__bf16)in[i];
}

// ---------------------------------------------------------------------------
// Transpose + convert: in [K,N] f32 -> out [N,K] bf16  (weights -> Bt layout)
// ---------------------------------------------------------------------------
__global__ void transpose_cvt_kernel(const float* __restrict__ in, __bf16* __restrict__ out,
                                     int K, int N) {
  __shared__ float t[32][33];
  int n0 = blockIdx.x * 32, k0 = blockIdx.y * 32;
  int tx = threadIdx.x, ty = threadIdx.y;  // block (32,8)
#pragma unroll
  for (int j = 0; j < 32; j += 8)
    t[ty + j][tx] = in[(size_t)(k0 + ty + j) * N + n0 + tx];
  __syncthreads();
#pragma unroll
  for (int j = 0; j < 32; j += 8)
    out[(size_t)(n0 + ty + j) * K + k0 + tx] = (__bf16)t[tx][ty + j];
}

// ---------------------------------------------------------------------------
// Per-(b,h) transpose of V: v [BH][S][DK] bf16 -> vt [BH][DK][S] bf16
// ---------------------------------------------------------------------------
__global__ void transpose_v_kernel(const __bf16* __restrict__ v, __bf16* __restrict__ vt) {
  __shared__ __bf16 t[32][33];
  int bh = blockIdx.z;
  int d0 = blockIdx.x * 32, s0 = blockIdx.y * 32;
  const __bf16* vb = v + (size_t)bh * SEQ * DK_;
  __bf16* vo = vt + (size_t)bh * DK_ * SEQ;
  int tx = threadIdx.x, ty = threadIdx.y;  // block (32,8)
#pragma unroll
  for (int j = 0; j < 32; j += 8)
    t[ty + j][tx] = vb[(size_t)(s0 + ty + j) * DK_ + d0 + tx];
  __syncthreads();
#pragma unroll
  for (int j = 0; j < 32; j += 8)
    vo[(size_t)(d0 + ty + j) * SEQ + s0 + tx] = t[tx][ty + j];
}

// ---------------------------------------------------------------------------
// WMMA GEMM: C[M,N] = A_bf16[M,K] @ Bt_bf16[N,K]^T + bias[N]
// Block tile 128x128, 256 threads = 8 waves (2 M-waves x 4 N-waves),
// wave tile 64x32 = 4x2 WMMA tiles, K-step 32, double-buffered LDS filled by
// the async global->LDS DMA path (one barrier + one asynccnt wait per slab;
// slab i+1's DMA overlaps slab i's WMMAs).
// Epilogue modes:
//   0 MF32    : Cf = val
//   1 MF32BF  : Cf = val, Cb = bf16(val)
//   2 MRELUBF : Cb = bf16(max(val,0))
//   3 MHEADS  : Cb laid out as [B,H,S,DK] bf16
// ---------------------------------------------------------------------------
template <int MODE>
__global__ void gemm_bf16_kernel(const __bf16* __restrict__ A, const __bf16* __restrict__ Bt,
                                 const float* __restrict__ bias, float* __restrict__ Cf,
                                 __bf16* __restrict__ Cb, int M, int N, int K) {
  __shared__ __bf16 lA[2][128 * 40];
  __shared__ __bf16 lB[2][128 * 40];

  const int tid = threadIdx.x;
  const int lane = tid & 31;
  const int wid = tid >> 5;
  const int wm = wid & 1;        // 0..1
  const int wn = wid >> 1;       // 0..3
  const int l16 = lane & 15;
  const int sel = lane >> 4;

  const int n0 = blockIdx.x * 128;
  const int m0 = blockIdx.y * 128;

  const int ldrow = tid >> 1;         // 0..127
  const int ldhalf = (tid & 1) * 16;  // 0 or 16 elements

  const __bf16* gA = A + (size_t)(m0 + ldrow) * K + ldhalf;
  const __bf16* gB = Bt + (size_t)(n0 + ldrow) * K + ldhalf;

  auto stage = [&](int k0, int buf) {
    __bf16* pa = &lA[buf][ldrow * 40 + ldhalf];
    __bf16* pb = &lB[buf][ldrow * 40 + ldhalf];
    async_copy_b128(gA + k0, pa);
    async_copy_b128(gA + k0 + 8, pa + 8);
    async_copy_b128(gB + k0, pb);
    async_copy_b128(gB + k0 + 8, pb + 8);
  };

  stage(0, 0);

  v8f acc[4][2] = {};

  int buf = 0;
  for (int k0 = 0; k0 < K; k0 += 32, buf ^= 1) {
    wait_async0();    // own async DMAs into lX[buf] have landed
    __syncthreads();  // ... and everyone else's; lX[buf^1] free for reuse
    if (k0 + 32 < K) stage(k0 + 32, buf ^ 1);

    v16bf bfrag[2];
#pragma unroll
    for (int ni = 0; ni < 2; ++ni)
      bfrag[ni] = frag_b(&lB[buf][(wn * 32 + ni * 16 + l16) * 40], sel);
#pragma unroll
    for (int mi = 0; mi < 4; ++mi) {
      v16bf afrag = frag_a(&lA[buf][(wm * 64 + mi * 16 + l16) * 40], sel);
#pragma unroll
      for (int ni = 0; ni < 2; ++ni)
        acc[mi][ni] = WMMA_BF16(afrag, bfrag[ni], acc[mi][ni]);
    }
  }

  // epilogue: C layout -> row = sel*8 + r, col = l16 within each 16x16 tile
#pragma unroll
  for (int mi = 0; mi < 4; ++mi) {
#pragma unroll
    for (int ni = 0; ni < 2; ++ni) {
      int n = n0 + wn * 32 + ni * 16 + l16;
      float bn = bias[n];
#pragma unroll
      for (int r = 0; r < 8; ++r) {
        int m = m0 + wm * 64 + mi * 16 + sel * 8 + r;
        float val = acc[mi][ni][r] + bn;
        if constexpr (MODE == 0) {
          Cf[(size_t)m * N + n] = val;
        } else if constexpr (MODE == 1) {
          Cf[(size_t)m * N + n] = val;
          Cb[(size_t)m * N + n] = (__bf16)val;
        } else if constexpr (MODE == 2) {
          Cb[(size_t)m * N + n] = (__bf16)fmaxf(val, 0.0f);
        } else {  // MHEADS: [B,H,S,DK]
          int b = m >> 11, s = m & (SEQ - 1);
          int h = n >> 6, d = n & (DK_ - 1);
          Cb[(((size_t)(b * NHEAD + h)) * SEQ + s) * DK_ + d] = (__bf16)val;
        }
      }
    }
  }
}

// ---------------------------------------------------------------------------
// Flash attention, LDS-cooperative version:
//   block = 8 waves (256 threads); each wave owns a 16-row q-tile
//   (128 q rows per block). K chunk (32x64) and Vt chunk (64x32) are staged
//   into double-buffered LDS via async global->LDS DMA; one barrier per chunk,
//   staging of chunk i+1 overlaps compute of chunk i. P is wave-private in LDS
//   (same-wave DS ops are in-order; only a compiler wave_barrier needed).
// ---------------------------------------------------------------------------
#define FA_WAVES 8
__global__ void flash_attn_kernel(const __bf16* __restrict__ Q, const __bf16* __restrict__ Kb,
                                  const __bf16* __restrict__ Vt, __bf16* __restrict__ ctx) {
  __shared__ __bf16 lK[2][32 * 72];           // K rows, stride 72 (conflict-free)
  __shared__ __bf16 lV[2][64 * 40];           // Vt rows, stride 40
  __shared__ __bf16 lP[FA_WAVES * 16 * 40];   // wave-private P tiles

  const int bh = blockIdx.y;                  // 0..31
  const int tid = threadIdx.x;                // 0..255
  const int lane = tid & 31;
  const int wid = tid >> 5;                   // 0..7
  const int l16 = lane & 15;
  const int sel = lane >> 4;
  const int q0 = blockIdx.x * (FA_WAVES * 16) + wid * 16;

  const __bf16* q = Q + (size_t)bh * SEQ * DK_;
  const __bf16* k = Kb + (size_t)bh * SEQ * DK_;
  const __bf16* vt = Vt + (size_t)bh * DK_ * SEQ;

  // cooperative staging: each thread DMAs one 16B chunk of K and one of Vt
  const int krow = tid >> 3, kcol = (tid & 7) * 8;   // 32 rows x 8 segs
  const int vrow = tid >> 2, vcol = (tid & 3) * 8;   // 64 rows x 4 segs

  auto stage = [&](int kc, int buf) {
    async_copy_b128(k + (size_t)(kc + krow) * DK_ + kcol, &lK[buf][krow * 72 + kcol]);
    async_copy_b128(vt + (size_t)vrow * SEQ + kc + vcol, &lV[buf][vrow * 40 + vcol]);
  };

  stage(0, 0);

  // Q fragments stay in registers (16 rows x 64 K)
  v16bf aq0, aq1;
  {
    const __bf16* qrow = q + (size_t)(q0 + l16) * DK_;
    aq0 = frag_a(qrow, sel);
    aq1 = frag_a(qrow + 32, sel);
  }

  float mrow[8], lrow[8];
  v8f acc[4] = {};
#pragma unroll
  for (int r = 0; r < 8; ++r) { mrow[r] = -1e30f; lrow[r] = 0.0f; }

  // 1/sqrt(DK) folded with log2(e): softmax runs in the exp2 domain
  const float SC = 0.125f * 1.44269504088896340736f;
  __bf16* myP = lP + wid * (16 * 40);

  int buf = 0;
  for (int kc = 0; kc < SEQ; kc += 32, buf ^= 1) {
    wait_async0();    // own DMAs into buf landed
    __syncthreads();  // everyone's landed; buf^1 free
    if (kc + 32 < SEQ) stage(kc + 32, buf ^ 1);

    const __bf16* lk = lK[buf];
    const __bf16* lv = lV[buf];

    // scores: two 16x16 kv tiles, K-dim = DK = 64 (two WMMA k-steps each)
    v8f s0 = {}, s1 = {};
    {
      const __bf16* kr0 = lk + l16 * 72;
      s0 = WMMA_BF16(aq0, frag_b(kr0, sel), s0);
      s0 = WMMA_BF16(aq1, frag_b(kr0 + 32, sel), s0);
      const __bf16* kr1 = lk + (16 + l16) * 72;
      s1 = WMMA_BF16(aq0, frag_b(kr1, sel), s1);
      s1 = WMMA_BF16(aq1, frag_b(kr1 + 32, sel), s1);
    }

    // online softmax (VALU + DPP + v_exp), store P bf16 into wave-private LDS
#pragma unroll
    for (int r = 0; r < 8; ++r) {
      float a0 = s0[r] * SC;
      float a1 = s1[r] * SC;
      float tm = redmax16(fmaxf(a0, a1));
      float mn = fmaxf(mrow[r], tm);
      float corr = __builtin_amdgcn_exp2f(mrow[r] - mn);
      mrow[r] = mn;
      float p0 = __builtin_amdgcn_exp2f(a0 - mn);
      float p1 = __builtin_amdgcn_exp2f(a1 - mn);
      float rs = redsum16(p0 + p1);
      lrow[r] = lrow[r] * corr + rs;
#pragma unroll
      for (int j = 0; j < 4; ++j) acc[j][r] *= corr;
      myP[(sel * 8 + r) * 40 + l16] = (__bf16)p0;
      myP[(sel * 8 + r) * 40 + 16 + l16] = (__bf16)p1;
    }
    __builtin_amdgcn_wave_barrier();  // keep DS order (same-wave DS in-order)

    // ctx += P @ V : K-dim = 32 kv, B fragments from staged Vt rows
    v16bf ap = frag_a(myP + l16 * 40, sel);
#pragma unroll
    for (int j = 0; j < 4; ++j)
      acc[j] = WMMA_BF16(ap, frag_b(lv + (j * 16 + l16) * 40, sel), acc[j]);
  }

  // finalize: ctx[token, h*64 + d] = acc / l
  const int b = bh >> 3, h = bh & 7;
#pragma unroll
  for (int r = 0; r < 8; ++r) {
    float inv = 1.0f / lrow[r];
    size_t tok = (size_t)(b * SEQ + q0 + sel * 8 + r);
#pragma unroll
    for (int j = 0; j < 4; ++j)
      ctx[tok * DMODEL + h * DK_ + j * 16 + l16] = (__bf16)(acc[j][r] * inv);
  }
}

// ---------------------------------------------------------------------------
// Residual + LayerNorm (in-place on h), fused bf16 output for next GEMM.
// One block per token row; 256 threads x 2 elements.
// ---------------------------------------------------------------------------
__global__ void resid_ln_kernel(const float* __restrict__ hin, const float* __restrict__ res,
                                const float* __restrict__ g, const float* __restrict__ bb,
                                float* __restrict__ hout, __bf16* __restrict__ hbf) {
  __shared__ float red[256];
  const int row = blockIdx.x, tid = threadIdx.x;
  const float* hr = hin + (size_t)row * DMODEL;
  const float* rr = res + (size_t)row * DMODEL;
  float x0 = hr[tid] + rr[tid];
  float x1 = hr[tid + 256] + rr[tid + 256];

  red[tid] = x0 + x1;
  __syncthreads();
  for (int s = 128; s > 0; s >>= 1) {
    if (tid < s) red[tid] += red[tid + s];
    __syncthreads();
  }
  float mu = red[0] * (1.0f / DMODEL);
  __syncthreads();

  float d0 = x0 - mu, d1 = x1 - mu;
  red[tid] = d0 * d0 + d1 * d1;
  __syncthreads();
  for (int s = 128; s > 0; s >>= 1) {
    if (tid < s) red[tid] += red[tid + s];
    __syncthreads();
  }
  float rstd = rsqrtf(red[0] * (1.0f / DMODEL) + 1e-5f);

  float y0 = d0 * rstd * g[tid] + bb[tid];
  float y1 = d1 * rstd * g[tid + 256] + bb[tid + 256];
  float* ho = hout + (size_t)row * DMODEL;
  __bf16* hb = hbf + (size_t)row * DMODEL;
  ho[tid] = y0;
  ho[tid + 256] = y1;
  hb[tid] = (__bf16)y0;
  hb[tid + 256] = (__bf16)y1;
}

// ---------------------------------------------------------------------------
// Mean-pool over sequence + final projection [512 x 10] (tiny, f32)
// ---------------------------------------------------------------------------
__global__ void pool_out_kernel(const float* __restrict__ h, const float* __restrict__ Wout,
                                const float* __restrict__ bout, float* __restrict__ out) {
  __shared__ float pooled[DMODEL];
  const int b = blockIdx.x, tid = threadIdx.x;  // 512 threads
  const float* hb = h + (size_t)b * SEQ * DMODEL;
  float s = 0.0f;
  for (int t = 0; t < SEQ; ++t) s += hb[(size_t)t * DMODEL + tid];
  pooled[tid] = s * (1.0f / SEQ);
  __syncthreads();
  if (tid < DOUT_) {
    float acc = bout[tid];
    for (int d = 0; d < DMODEL; ++d) acc += pooled[d] * Wout[d * DOUT_ + tid];
    out[b * DOUT_ + tid] = acc;
  }
}

// ---------------------------------------------------------------------------
// Host orchestration
// ---------------------------------------------------------------------------
extern "C" void kernel_launch(void* const* d_in, const int* in_sizes, int n_in,
                              void* d_out, int out_size, void* d_ws, size_t ws_size,
                              hipStream_t stream) {
  (void)in_sizes; (void)n_in; (void)out_size; (void)ws_size;
  const float* x    = (const float*)d_in[0];
  const float* Wp   = (const float*)d_in[1];
  const float* bp   = (const float*)d_in[2];
  const float* Wq   = (const float*)d_in[3];
  const float* bq   = (const float*)d_in[4];
  const float* Wk   = (const float*)d_in[5];
  const float* bk   = (const float*)d_in[6];
  const float* Wv   = (const float*)d_in[7];
  const float* bv   = (const float*)d_in[8];
  const float* Wo   = (const float*)d_in[9];
  const float* bo   = (const float*)d_in[10];
  const float* ln_g = (const float*)d_in[11];
  const float* ln_b = (const float*)d_in[12];
  const float* W1   = (const float*)d_in[13];
  const float* b1   = (const float*)d_in[14];
  const float* W2   = (const float*)d_in[15];
  const float* b2   = (const float*)d_in[16];
  const float* Wout = (const float*)d_in[17];
  const float* bout = (const float*)d_in[18];
  float* out = (float*)d_out;

  char* ws = (char*)d_ws;
  size_t off = 0;
  auto alloc = [&](size_t bytes) -> char* {
    char* p = ws + off;
    off += (bytes + 255) & ~(size_t)255;
    return p;
  };

  __bf16* xbf  = (__bf16*)alloc((size_t)NTOK * DIN_ * 2);
  float*  hf   = (float*)alloc((size_t)NTOK * DMODEL * 4);
  __bf16* hbf  = (__bf16*)alloc((size_t)NTOK * DMODEL * 2);
  float*  tmp  = (float*)alloc((size_t)NTOK * DMODEL * 4);
  __bf16* qb   = (__bf16*)alloc((size_t)NTOK * DMODEL * 2);
  __bf16* kb   = (__bf16*)alloc((size_t)NTOK * DMODEL * 2);
  __bf16* vb   = (__bf16*)alloc((size_t)NTOK * DMODEL * 2);
  __bf16* vtb  = (__bf16*)alloc((size_t)NTOK * DMODEL * 2);
  __bf16* ctxb = (__bf16*)alloc((size_t)NTOK * DMODEL * 2);
  __bf16* midbf= (__bf16*)alloc((size_t)NTOK * DFFN * 2);
  __bf16* WpT  = (__bf16*)alloc((size_t)DMODEL * DIN_ * 2);
  __bf16* WqT  = (__bf16*)alloc((size_t)LYR * DMODEL * DMODEL * 2);
  __bf16* WkT  = (__bf16*)alloc((size_t)LYR * DMODEL * DMODEL * 2);
  __bf16* WvT  = (__bf16*)alloc((size_t)LYR * DMODEL * DMODEL * 2);
  __bf16* WoT  = (__bf16*)alloc((size_t)LYR * DMODEL * DMODEL * 2);
  __bf16* W1T  = (__bf16*)alloc((size_t)LYR * DFFN * DMODEL * 2);
  __bf16* W2T  = (__bf16*)alloc((size_t)LYR * DMODEL * DFFN * 2);

  const dim3 tblk(32, 8);

  // x -> bf16
  cvt_bf16_kernel<<<(NTOK * DIN_ + 255) / 256, 256, 0, stream>>>(x, xbf, NTOK * DIN_);

  // weights -> transposed bf16 [N,K]
  transpose_cvt_kernel<<<dim3(DMODEL / 32, DIN_ / 32), tblk, 0, stream>>>(Wp, WpT, DIN_, DMODEL);
  for (int i = 0; i < LYR; ++i) {
    size_t wdd = (size_t)i * DMODEL * DMODEL;
    transpose_cvt_kernel<<<dim3(16, 16), tblk, 0, stream>>>(Wq + wdd, WqT + wdd, DMODEL, DMODEL);
    transpose_cvt_kernel<<<dim3(16, 16), tblk, 0, stream>>>(Wk + wdd, WkT + wdd, DMODEL, DMODEL);
    transpose_cvt_kernel<<<dim3(16, 16), tblk, 0, stream>>>(Wv + wdd, WvT + wdd, DMODEL, DMODEL);
    transpose_cvt_kernel<<<dim3(16, 16), tblk, 0, stream>>>(Wo + wdd, WoT + wdd, DMODEL, DMODEL);
    size_t wdf = (size_t)i * DMODEL * DFFN;
    transpose_cvt_kernel<<<dim3(DFFN / 32, DMODEL / 32), tblk, 0, stream>>>(W1 + wdf, W1T + wdf, DMODEL, DFFN);
    transpose_cvt_kernel<<<dim3(DMODEL / 32, DFFN / 32), tblk, 0, stream>>>(W2 + wdf, W2T + wdf, DFFN, DMODEL);
  }

  // input projection: h = x @ Wp + bp  (write f32 + bf16)
  gemm_bf16_kernel<1><<<dim3(DMODEL / 128, NTOK / 128), 256, 0, stream>>>(
      xbf, WpT, bp, hf, hbf, NTOK, DMODEL, DIN_);

  for (int i = 0; i < LYR; ++i) {
    size_t wdd = (size_t)i * DMODEL * DMODEL;
    size_t wdf = (size_t)i * DMODEL * DFFN;
    const dim3 g512(DMODEL / 128, NTOK / 128);

    // Q,K,V projections -> [B,H,S,DK] bf16
    gemm_bf16_kernel<3><<<g512, 256, 0, stream>>>(hbf, WqT + wdd, bq + i * DMODEL, nullptr, qb,
                                                  NTOK, DMODEL, DMODEL);
    gemm_bf16_kernel<3><<<g512, 256, 0, stream>>>(hbf, WkT + wdd, bk + i * DMODEL, nullptr, kb,
                                                  NTOK, DMODEL, DMODEL);
    gemm_bf16_kernel<3><<<g512, 256, 0, stream>>>(hbf, WvT + wdd, bv + i * DMODEL, nullptr, vb,
                                                  NTOK, DMODEL, DMODEL);
    // V -> Vt
    transpose_v_kernel<<<dim3(DK_ / 32, SEQ / 32, NBATCH * NHEAD), tblk, 0, stream>>>(vb, vtb);
    // attention (8 waves/block, 128 q-rows per block)
    flash_attn_kernel<<<dim3(SEQ / (FA_WAVES * 16), NBATCH * NHEAD), FA_WAVES * 32, 0, stream>>>(
        qb, kb, vtb, ctxb);
    // output projection
    gemm_bf16_kernel<0><<<g512, 256, 0, stream>>>(ctxb, WoT + wdd, bo + i * DMODEL, tmp, nullptr,
                                                  NTOK, DMODEL, DMODEL);
    // h = LN(h + attn_out)
    resid_ln_kernel<<<NTOK, 256, 0, stream>>>(hf, tmp, ln_g + i * DMODEL, ln_b + i * DMODEL, hf, hbf);
    // FFN1 (ReLU, bf16 out)
    gemm_bf16_kernel<2><<<dim3(DFFN / 128, NTOK / 128), 256, 0, stream>>>(
        hbf, W1T + wdf, b1 + i * DFFN, nullptr, midbf, NTOK, DFFN, DMODEL);
    // FFN2 (f32 out)
    gemm_bf16_kernel<0><<<g512, 256, 0, stream>>>(midbf, W2T + wdf, b2 + i * DMODEL, tmp, nullptr,
                                                  NTOK, DMODEL, DFFN);
    // h = LN(h + ffn_out)
    resid_ln_kernel<<<NTOK, 256, 0, stream>>>(hf, tmp, ln_g + i * DMODEL, ln_b + i * DMODEL, hf, hbf);
  }

  // mean pool + output head
  pool_out_kernel<<<NBATCH, DMODEL, 0, stream>>>(hf, Wout, bout, out);
}
